// VectorQuantizer_83193516524131
// MI455X (gfx1250) — matrix operand order repaired and checked by reading, not compiled, not run on hardware
//
#include <hip/hip_runtime.h>
#include <hip/hip_bf16.h>

typedef __attribute__((ext_vector_type(16))) __bf16 v16bf;
typedef __attribute__((ext_vector_type(8)))  float  v8f;

#define B_ROWS   65536
#define K_CODES  8192
#define DIMS     64
#define CHUNK    128         // codebook rows staged in LDS per iteration (double buffered)
#define NCHUNK   (K_CODES / CHUNK)
#define ROW_PAD  72          // bf16 elements per LDS row (64 + 8 pad) -> 144B stride, 36-dword bank stride
#define WG_ROWS  128         // 8 waves * 16 rows

// ---- CDNA5 async global->LDS path (ASYNCcnt), with safe fallback ----------
#if defined(__has_builtin)
#  if __has_builtin(__builtin_amdgcn_global_load_async_to_lds_b128)
#    define HAVE_ASYNC_B128 1
#  else
#    define HAVE_ASYNC_B128 0
#  endif
#  if __has_builtin(__builtin_amdgcn_global_load_async_to_lds_b32)
#    define HAVE_ASYNC_B32 1
#  else
#    define HAVE_ASYNC_B32 0
#  endif
#else
#  define HAVE_ASYNC_B128 0
#  define HAVE_ASYNC_B32  0
#endif

// exact pointee types per hipcc diagnostic:
//   b128: 'int __attribute__((vector_size(16))) __device__ *'
//   b32 : 'int __device__ *'
typedef int vsi4 __attribute__((vector_size(16)));
typedef __attribute__((address_space(1))) vsi4* g_vsi4_p;
typedef __attribute__((address_space(3))) vsi4* l_vsi4_p;
typedef __attribute__((address_space(1))) int*  g_i32_p;
typedef __attribute__((address_space(3))) int*  l_i32_p;

static __device__ __forceinline__ void copy16_g2l(const void* g, void* l) {
#if HAVE_ASYNC_B128
    __builtin_amdgcn_global_load_async_to_lds_b128((g_vsi4_p)g, (l_vsi4_p)l, 0, 0);
#else
    *(uint4*)l = *(const uint4*)g;
#endif
}

static __device__ __forceinline__ void copy4_g2l(const void* g, void* l) {
#if HAVE_ASYNC_B32
    __builtin_amdgcn_global_load_async_to_lds_b32((g_i32_p)g, (l_i32_p)l, 0, 0);
#else
    *(unsigned int*)l = *(const unsigned int*)g;
#endif
}

static __device__ __forceinline__ void wait_async_copies() {
#if HAVE_ASYNC_B128
#  if defined(__has_builtin) && __has_builtin(__builtin_amdgcn_s_wait_asynccnt)
    __builtin_amdgcn_s_wait_asynccnt(0);
#  else
    asm volatile("s_wait_asynccnt 0x0" ::: "memory");
#  endif
#endif
}

// round-to-nearest-even f32 -> bf16 (bit pattern)
static __device__ __forceinline__ unsigned short f2bf(float f) {
    unsigned int u = __float_as_uint(f);
    u = (u + 0x7FFFu + ((u >> 16) & 1u)) >> 16;
    return (unsigned short)u;
}

union FragA { unsigned short s[16]; v16bf v; };

// ---------------------------------------------------------------------------
// Kernel 1: codebook -> bf16 copy (workspace) + per-code squared norms
// ---------------------------------------------------------------------------
__global__ __launch_bounds__(256) void vq_prep(const float* __restrict__ cb,
                                               unsigned short* __restrict__ cb_bf,
                                               float* __restrict__ cb_norm) {
    int row = blockIdx.x * 256 + threadIdx.x;          // 8192 rows total
    const float4* src = (const float4*)(cb + (size_t)row * DIMS);
    uint4* dst = (uint4*)(cb_bf + (size_t)row * DIMS);
    float nrm = 0.f;
#pragma unroll
    for (int g = 0; g < 8; ++g) {
        float4 a = src[2 * g + 0];
        float4 b = src[2 * g + 1];
        nrm += a.x * a.x + a.y * a.y + a.z * a.z + a.w * a.w;
        nrm += b.x * b.x + b.y * b.y + b.z * b.z + b.w * b.w;
        union { unsigned short h[8]; uint4 u; } pk;
        pk.h[0] = f2bf(a.x); pk.h[1] = f2bf(a.y); pk.h[2] = f2bf(a.z); pk.h[3] = f2bf(a.w);
        pk.h[4] = f2bf(b.x); pk.h[5] = f2bf(b.y); pk.h[6] = f2bf(b.z); pk.h[7] = f2bf(b.w);
        dst[g] = pk.u;
    }
    cb_norm[row] = nrm;
}

// ---------------------------------------------------------------------------
// Kernel 2: fused distance-GEMM (bf16 WMMA) + argmin + gather + loss partials
// Double-buffered LDS staging via CDNA5 async global->LDS copies.
// ---------------------------------------------------------------------------
__global__ __launch_bounds__(256) void vq_main(const float* __restrict__ x,
                                               const float* __restrict__ cb,
                                               const unsigned short* __restrict__ cb_bf,
                                               const float* __restrict__ cb_norm,
                                               float* __restrict__ out_q,
                                               int* __restrict__ out_idx,
                                               float* __restrict__ partial) {
    __shared__ unsigned short sB[2][CHUNK * ROW_PAD]; // 2 x 18 KB staged bf16 codebook chunks
    __shared__ float sN[2][CHUNK];                    // staged norms
    __shared__ int   sIdx[WG_ROWS];                   // per-row argmin for gather phase
    __shared__ float sRed[256];                       // loss reduction

    const int tid  = threadIdx.x;
    const int wave = tid >> 5;                        // 0..7
    const int lane = tid & 31;
    const int col  = lane & 15;                       // lane's fixed N column (mod 16)
    const int half = lane >> 4;                       // 0: M/K low half, 1: high half
    const int m_base = blockIdx.x * WG_ROWS + wave * 16;

    // ---- Build A fragments (this wave's 16 input rows, bf16) once. ----
    // A k order used here (B matches): lane half h, element i -> k = h*16+i
    FragA a0, a1;
    {
        const float* xr = x + (size_t)(m_base + col) * DIMS;
#pragma unroll
        for (int f = 0; f < 2; ++f) {
            int fbase = f * 32 + half * 16;
            float4 p0 = *(const float4*)(xr + fbase + 0);
            float4 p1 = *(const float4*)(xr + fbase + 4);
            float4 q0 = *(const float4*)(xr + fbase + 8);
            float4 q1 = *(const float4*)(xr + fbase + 12);
            unsigned short* h = f ? a1.s : a0.s;
            h[0]  = f2bf(p0.x); h[1]  = f2bf(p0.y); h[2]  = f2bf(p0.z); h[3]  = f2bf(p0.w);
            h[4]  = f2bf(p1.x); h[5]  = f2bf(p1.y); h[6]  = f2bf(p1.z); h[7]  = f2bf(p1.w);
            h[8]  = f2bf(q0.x); h[9]  = f2bf(q0.y); h[10] = f2bf(q0.z); h[11] = f2bf(q0.w);
            h[12] = f2bf(q1.x); h[13] = f2bf(q1.y); h[14] = f2bf(q1.z); h[15] = f2bf(q1.w);
        }
    }

    float minv[8];
    int   mini[8];
#pragma unroll
    for (int r = 0; r < 8; ++r) { minv[r] = 3.4e38f; mini[r] = 0; }

    // ---- Stage helper: thread t copies half of codebook row (t>>1). ----
    const int srow = tid >> 1;            // 0..127
    const int shlf = tid & 1;             // 0/1 -> 64B halves
    // prologue: stage chunk 0 into buffer 0
    {
        const unsigned short* src = cb_bf + (size_t)srow * DIMS + shlf * 32;
        unsigned short* dst = &sB[0][srow * ROW_PAD + shlf * 32];
#pragma unroll
        for (int g = 0; g < 4; ++g) copy16_g2l(src + g * 8, dst + g * 8);
        if (tid < CHUNK) copy4_g2l(cb_norm + tid, &sN[0][tid]);
    }

    // ---- Sweep the codebook: stage chunk nc+1 while computing chunk nc. ----
    for (int nc = 0; nc < NCHUNK; ++nc) {
        const int buf = nc & 1;
        wait_async_copies();              // chunk nc resident in sB[buf]
        __syncthreads();                  // ...and all waves done with sB[buf^1]

        if (nc + 1 < NCHUNK) {            // prefetch next chunk into other buffer
            const unsigned short* src =
                cb_bf + (size_t)((nc + 1) * CHUNK + srow) * DIMS + shlf * 32;
            unsigned short* dst = &sB[buf ^ 1][srow * ROW_PAD + shlf * 32];
#pragma unroll
            for (int g = 0; g < 4; ++g) copy16_g2l(src + g * 8, dst + g * 8);
            if (tid < CHUNK) copy4_g2l(cb_norm + (nc + 1) * CHUNK + tid, &sN[buf ^ 1][tid]);
        }

        const unsigned short* bbase = sB[buf];
#pragma unroll 2
        for (int t = 0; t < CHUNK / 16; ++t) {
            int nl = t * 16 + col;
            // B fragment: lane holds column nl, K consecutive per half.
            // frag f covers d = f*32 + half*16 + (0..15)  -> contiguous 32B in LDS
            FragA b0, b1;
            {
                const uint4* p0 = (const uint4*)(bbase + nl * ROW_PAD + half * 16);
                const uint4* p1 = (const uint4*)(bbase + nl * ROW_PAD + 32 + half * 16);
                ((uint4*)b0.s)[0] = p0[0]; ((uint4*)b0.s)[1] = p0[1];
                ((uint4*)b1.s)[0] = p1[0]; ((uint4*)b1.s)[1] = p1[1];
            }
            v8f c = {};
            c = __builtin_amdgcn_wmma_f32_16x16x32_bf16(false, a0.v, false, b0.v,
                                                        (short)0, c, false, false);
            c = __builtin_amdgcn_wmma_f32_16x16x32_bf16(false, a1.v, false, b1.v,
                                                        (short)0, c, false, false);
            float en = sN[buf][nl];
            int   ng = nc * CHUNK + nl;   // global code index for this lane
#pragma unroll
            for (int r = 0; r < 8; ++r) {
                float dist = __builtin_fmaf(-2.f, c[r], en);  // ||e||^2 - 2 x.e
                bool lt = dist < minv[r];
                minv[r] = lt ? dist : minv[r];
                mini[r] = lt ? ng   : mini[r];
            }
        }
    }

    // ---- Cross-lane argmin within each 16-lane half (xor butterfly). ----
#pragma unroll
    for (int r = 0; r < 8; ++r) {
        float v = minv[r];
        int   i = mini[r];
#pragma unroll
        for (int off = 1; off < 16; off <<= 1) {
            float ov = __shfl_xor(v, off, 32);
            int   oi = __shfl_xor(i, off, 32);
            if (ov < v || (ov == v && oi < i)) { v = ov; i = oi; }
        }
        if (col == 0) {                   // lanes 0 & 16: rows r / r+8
            int ml = wave * 16 + half * 8 + r;
            sIdx[ml] = i;
            out_idx[blockIdx.x * WG_ROWS + ml] = i;
        }
    }
    __syncthreads();

    // ---- Gather quantized rows (exact f32) + loss partial. ----
    float lsum = 0.f;
    {
        int rl   = tid >> 1;              // 0..127
        int off  = (tid & 1) * 32;
        int grow = blockIdx.x * WG_ROWS + rl;
        int idx  = sIdx[rl];
        const float4* qs = (const float4*)(cb + (size_t)idx * DIMS + off);
        const float4* xs = (const float4*)(x + (size_t)grow * DIMS + off);
        float4* qd = (float4*)(out_q + (size_t)grow * DIMS + off);
#pragma unroll
        for (int g = 0; g < 8; ++g) {
            float4 q = qs[g];
            float4 v = xs[g];
            qd[g] = q;
            float dx = v.x - q.x, dy = v.y - q.y, dz = v.z - q.z, dw = v.w - q.w;
            lsum += dx * dx + dy * dy + dz * dz + dw * dw;
        }
    }
    sRed[tid] = lsum;
    __syncthreads();
#pragma unroll
    for (int s = 128; s > 0; s >>= 1) {
        if (tid < s) sRed[tid] += sRed[tid + s];
        __syncthreads();
    }
    if (tid == 0) partial[blockIdx.x] = sRed[0];
}

// ---------------------------------------------------------------------------
// Kernel 3: deterministic fixed-order reduction of 512 partials -> vq_loss
// ---------------------------------------------------------------------------
__global__ __launch_bounds__(256) void vq_finish(const float* __restrict__ partial,
                                                 float* __restrict__ loss) {
    __shared__ float s[256];
    int t = threadIdx.x;
    s[t] = partial[t] + partial[t + 256];
    __syncthreads();
#pragma unroll
    for (int k = 128; k > 0; k >>= 1) {
        if (t < k) s[t] += s[t + k];
        __syncthreads();
    }
    if (t == 0)
        loss[0] = 1.25f * s[0] / ((float)B_ROWS * (float)DIMS);
}

// ---------------------------------------------------------------------------
extern "C" void kernel_launch(void* const* d_in, const int* in_sizes, int n_in,
                              void* d_out, int out_size, void* d_ws, size_t ws_size,
                              hipStream_t stream) {
    (void)in_sizes; (void)n_in; (void)out_size; (void)ws_size;
    const float* x  = (const float*)d_in[0];          // [65536, 64]
    const float* cb = (const float*)d_in[1];          // [8192, 64]

    float* out      = (float*)d_out;
    float* out_q    = out;                            // [65536*64] quantized_st
    float* out_loss = out + (size_t)B_ROWS * DIMS;    // [1] vq_loss
    int*   out_idx  = (int*)(out + (size_t)B_ROWS * DIMS + 1); // [65536] indices (int32)

    char* ws = (char*)d_ws;
    unsigned short* cb_bf = (unsigned short*)ws;                        // 1 MB
    float* cb_norm = (float*)(ws + (size_t)K_CODES * DIMS * 2);         // 32 KB
    float* partial = (float*)(ws + (size_t)K_CODES * DIMS * 2 + K_CODES * 4); // 2 KB

    vq_prep  <<<K_CODES / 256, 256, 0, stream>>>(cb, cb_bf, cb_norm);
    vq_main  <<<B_ROWS / WG_ROWS, 256, 0, stream>>>(x, cb, cb_bf, cb_norm,
                                                    out_q, out_idx, partial);
    vq_finish<<<1, 256, 0, stream>>>(partial, out_loss);
}
